// CharRNN_4973572129487
// MI455X (gfx1250) — matrix operand order repaired and checked by reading, not compiled, not run on hardware
//
#include <hip/hip_runtime.h>
#include <hip/hip_bf16.h>
#include <stdint.h>

// ---------------- problem constants (from reference) ----------------
#define T_STEPS 256
#define BATCH   64
#define XDIM    512
#define HDIM    2048
#define GDIM    (4 * HDIM)        // 8192 gate columns
#define KDIM    (XDIM + HDIM)     // 2560 fused K (x | h)
#define KC      128               // K chunk staged in LDS per iteration
#define NCH     (KDIM / KC)       // 20 chunks (4 x-chunks, 16 h-chunks)
#define PITCH   136               // LDS panel row pitch in halves (128 + 8 pad -> conflict-free b128)
#define PANEL_HALVES (64 * PITCH) // one panel buffer

typedef __attribute__((ext_vector_type(16))) __bf16 bf16x16;
typedef __attribute__((ext_vector_type(8)))  float  floatx8;

union Frag16 { bf16x16 m; uint4 q[2]; };

__device__ __forceinline__ unsigned short f32_to_bf16_rne(float x) {
  unsigned int u = __float_as_uint(x);
  u += 0x7FFFu + ((u >> 16) & 1u);   // round-to-nearest-even
  return (unsigned short)(u >> 16);
}

__device__ __forceinline__ float sigm(float x) { return 1.0f / (1.0f + __expf(-x)); }

// CDNA5 async copy: global -> LDS, tracked by ASYNCcnt (no VGPR round-trip)
__device__ __forceinline__ void async_copy_b128(unsigned lds_off, const unsigned short* gp) {
  asm volatile("global_load_async_to_lds_b128 %0, %1, off"
               :: "v"(lds_off), "v"(gp) : "memory");
}
__device__ __forceinline__ void wait_async0() {
  asm volatile("s_wait_asynccnt 0" ::: "memory");
}

// ---------------- prep kernels ----------------

// Wc[n][k] : k<XDIM -> w_ih[n][k], else w_hh[n][k-XDIM], converted to bf16
__global__ void prep_weights(const float* __restrict__ w_ih,
                             const float* __restrict__ w_hh,
                             unsigned short* __restrict__ Wc) {
  const size_t total = (size_t)GDIM * KDIM;
  for (size_t i = (size_t)blockIdx.x * blockDim.x + threadIdx.x; i < total;
       i += (size_t)gridDim.x * blockDim.x) {
    int k = (int)(i % KDIM);
    int n = (int)(i / KDIM);
    float v = (k < XDIM) ? w_ih[(size_t)n * XDIM + k]
                         : w_hh[(size_t)n * HDIM + (k - XDIM)];
    Wc[i] = f32_to_bf16_rne(v);
  }
}

__global__ void prep_input(const float* __restrict__ in, unsigned short* __restrict__ xb) {
  const size_t total = (size_t)T_STEPS * BATCH * XDIM;
  for (size_t i = (size_t)blockIdx.x * blockDim.x + threadIdx.x; i < total;
       i += (size_t)gridDim.x * blockDim.x)
    xb[i] = f32_to_bf16_rne(in[i]);
}

__global__ void prep_state(const float* __restrict__ h0, const float* __restrict__ c0,
                           const float* __restrict__ b_ih, const float* __restrict__ b_hh,
                           unsigned short* __restrict__ hb0, float* __restrict__ cf,
                           float* __restrict__ bsum) {
  int i = blockIdx.x * blockDim.x + threadIdx.x;
  if (i < BATCH * HDIM) {
    hb0[i] = f32_to_bf16_rne(h0[i]);
    cf[i]  = c0[i];
  }
  if (i < GDIM) bsum[i] = b_ih[i] + b_hh[i];
}

// ---------------- recurrent step: fused GEMM (WMMA bf16) + LSTM pointwise ----------------
// Grid: HDIM/32 = 64 blocks, 256 threads (8 waves).
// Block owns h-columns [nblk, nblk+32). Wave w: gate g = w>>1, n-tile nt = w&1.
// Each wave accumulates all M=64 rows (4 m-tiles) for its 16 gate columns.
// A panel (x_t | h) is double-buffered in LDS via GLOBAL_LOAD_ASYNC_TO_LDS_B128 so the
// copy of chunk ch+1 overlaps WMMA compute on chunk ch.
__global__ __launch_bounds__(256) void lstm_step(
    const unsigned short* __restrict__ xb,    // [T,B,X] bf16
    const unsigned short* __restrict__ Wc,    // [GDIM,KDIM] bf16
    const float* __restrict__ bsum,           // [GDIM]
    const unsigned short* __restrict__ hin,   // [B,H] bf16 (prev h)
    unsigned short* __restrict__ hout,        // [B,H] bf16 (next h)
    float* __restrict__ hf,                   // [B,H] f32 (latest h)
    float* __restrict__ cf,                   // [B,H] f32 (cell, in/out)
    float* __restrict__ ys,                   // [T,B,H] f32
    int t) {
  // Double-buffered A panel; gate tiles alias the same LDS (live only after K-loop).
  __shared__ __align__(16) unsigned char smem[2 * PANEL_HALVES * 2];  // 34816 B
  unsigned short (*panel)[PANEL_HALVES] =
      reinterpret_cast<unsigned short (*)[PANEL_HALVES]>(smem);
  float* gates = reinterpret_cast<float*>(smem);   // [4][64][32] = 32768 B, reused after GEMM

  const int tid  = threadIdx.x;
  const int lane = tid & 31;
  const int w    = tid >> 5;
  const int g    = w >> 1;          // gate 0..3 (i,f,g,o)
  const int nt   = w & 1;           // 16-col tile within block
  const int nblk = blockIdx.x * 32;

  const int r  = lane & 15;
  const int hi = lane >> 4;

  // B operand: B[k][n] = Wc[n][k]; lane holds column n = wrow, K rows hi*8..+7 and 16+hi*8..+7 per frag
  const int wrow = g * HDIM + nblk + nt * 16 + r;
  const unsigned short* wbase = Wc + (size_t)wrow * KDIM + hi * 8;

  const unsigned short* xt = xb + (size_t)t * BATCH * XDIM;

  // issue async loads for chunk ch into panel[buf] (each thread: 4 x b128)
  auto issue_chunk = [&](int ch, int buf) {
    const unsigned short* src;
    int ld, koff;
    if (ch < XDIM / KC) { src = xt;  ld = XDIM; koff = ch * KC; }
    else                { src = hin; ld = HDIM; koff = (ch - XDIM / KC) * KC; }
    #pragma unroll
    for (int i = 0; i < 4; ++i) {
      int q   = tid + i * 256;      // 1024 uint4 = 64 rows x 16 uint4
      int row = q >> 4;
      int c8  = q & 15;
      unsigned lds_off =
          (unsigned)(uintptr_t)&panel[buf][row * PITCH + c8 * 8];
      const unsigned short* gp = src + (size_t)row * ld + koff + c8 * 8;
      async_copy_b128(lds_off, gp);
    }
  };

  floatx8 acc[4] = {};   // C tiles for m-tiles 0..3 (batch rows 0..63)

  issue_chunk(0, 0);
  wait_async0();
  __syncthreads();

  #pragma unroll 1
  for (int ch = 0; ch < NCH; ++ch) {
    if (ch + 1 < NCH) {
      issue_chunk(ch + 1, (ch + 1) & 1);              // overlap copy with compute
      __builtin_prefetch(wbase + (size_t)(ch + 1) * KC, 0, 0);  // warm next weight chunk
    }

    const unsigned short* pan = panel[ch & 1];
    const int kg = ch * KC;
    #pragma unroll
    for (int kk = 0; kk < KC; kk += 32) {
      Frag16 bfrag;  // 32x16 bf16 B tile: lanes 0-15 K=kk+0..7,16..23, lanes 16-31 K=kk+8..15,24..31
      const unsigned short* bp = wbase + kg + kk;
      bfrag.q[0] = *(const uint4*)(bp);
      bfrag.q[1] = *(const uint4*)(bp + 16);
      // batch all A-fragment LDS loads first so they clause and waits can stagger
      Frag16 af[4];
      #pragma unroll
      for (int mt = 0; mt < 4; ++mt) {
        const unsigned short* ap = &pan[(mt * 16 + r) * PITCH + kk + hi * 8];
        af[mt].q[0] = *(const uint4*)(ap);
        af[mt].q[1] = *(const uint4*)(ap + 16);
      }
      #pragma unroll
      for (int mt = 0; mt < 4; ++mt)
        acc[mt] = __builtin_amdgcn_wmma_f32_16x16x32_bf16(
            false, af[mt].m, false, bfrag.m, (short)0, acc[mt], false, false);
    }

    wait_async0();       // next panel resident (per-wave ASYNCcnt)
    __syncthreads();     // all waves done computing / async-arrived
  }

  // publish gate tiles to LDS (aliases the dead panel buffers)
  #pragma unroll
  for (int mt = 0; mt < 4; ++mt) {
    #pragma unroll
    for (int v = 0; v < 8; ++v) {
      int m = mt * 16 + hi * 8 + v;          // C layout: vgpr v -> M = v + hi*8
      gates[(g * 64 + m) * 32 + nt * 16 + r] = acc[mt][v];
    }
  }
  __syncthreads();

  // fused LSTM pointwise over this block's 64x32 outputs
  for (int e = tid; e < 64 * 32; e += 256) {
    int m  = e >> 5;
    int nl = e & 31;
    int n  = nblk + nl;
    float gi = gates[(0 * 64 + m) * 32 + nl] + bsum[0 * HDIM + n];
    float gv = gates[(1 * 64 + m) * 32 + nl] + bsum[1 * HDIM + n];
    float gg = gates[(2 * 64 + m) * 32 + nl] + bsum[2 * HDIM + n];
    float go = gates[(3 * 64 + m) * 32 + nl] + bsum[3 * HDIM + n];
    float iv = sigm(gi);
    float fv = sigm(gv);
    float cg = tanhf(gg);
    float ov = sigm(go);
    size_t idx = (size_t)m * HDIM + n;
    float cn = fv * cf[idx] + iv * cg;
    float hn = ov * tanhf(cn);
    cf[idx]   = cn;
    hf[idx]   = hn;
    hout[idx] = f32_to_bf16_rne(hn);
    // ys is a 134 MB stream: non-temporal so it doesn't evict the L2-resident weights
    __builtin_nontemporal_store(hn, ys + (size_t)t * BATCH * HDIM + idx);
  }
}

// ---------------- batched softmax over all T*B rows of ys, in place ----------------
__global__ __launch_bounds__(256) void softmax_rows(float* __restrict__ ys) {
  __shared__ float red[256];
  const int tid = threadIdx.x;
  float* row = ys + (size_t)blockIdx.x * HDIM;
  float v[8];
  float lmax = -3.402823466e38f;
  #pragma unroll
  for (int i = 0; i < 8; ++i) { v[i] = row[tid + i * 256]; lmax = fmaxf(lmax, v[i]); }
  red[tid] = lmax;
  __syncthreads();
  for (int s = 128; s > 0; s >>= 1) {
    if (tid < s) red[tid] = fmaxf(red[tid], red[tid + s]);
    __syncthreads();
  }
  float rmax = red[0];
  __syncthreads();
  float lsum = 0.f;
  #pragma unroll
  for (int i = 0; i < 8; ++i) { v[i] = __expf(v[i] - rmax); lsum += v[i]; }
  red[tid] = lsum;
  __syncthreads();
  for (int s = 128; s > 0; s >>= 1) {
    if (tid < s) red[tid] += red[tid + s];
    __syncthreads();
  }
  float inv = 1.0f / red[0];
  #pragma unroll
  for (int i = 0; i < 8; ++i)
    __builtin_nontemporal_store(v[i] * inv, row + tid + i * 256);
}

__global__ void write_final(const float* __restrict__ hf, const float* __restrict__ cf,
                            float* __restrict__ out) {
  int i = blockIdx.x * blockDim.x + threadIdx.x;
  if (i < BATCH * HDIM) {
    out[i]                = hf[i];   // h_T
    out[BATCH * HDIM + i] = cf[i];   // c_T
  }
}

// ---------------- host launcher ----------------
extern "C" void kernel_launch(void* const* d_in, const int* in_sizes, int n_in,
                              void* d_out, int out_size, void* d_ws, size_t ws_size,
                              hipStream_t stream) {
  const float* input = (const float*)d_in[0];  // [T,B,X]
  const float* h0    = (const float*)d_in[1];  // [B,H]
  const float* c0    = (const float*)d_in[2];  // [B,H]
  const float* w_ih  = (const float*)d_in[3];  // [4H,X]
  const float* w_hh  = (const float*)d_in[4];  // [4H,H]
  const float* b_ih  = (const float*)d_in[5];  // [4H]
  const float* b_hh  = (const float*)d_in[6];  // [4H]
  float* out = (float*)d_out;                  // ys | h_T | c_T

  // workspace carve-out (~60 MB), all chunks 256B-aligned by construction
  char* p = (char*)d_ws;
  unsigned short* Wc  = (unsigned short*)p; p += (size_t)GDIM * KDIM * 2;            // 41.9 MB
  unsigned short* xb  = (unsigned short*)p; p += (size_t)T_STEPS * BATCH * XDIM * 2; // 16.8 MB
  unsigned short* hb0 = (unsigned short*)p; p += (size_t)BATCH * HDIM * 2;
  unsigned short* hb1 = (unsigned short*)p; p += (size_t)BATCH * HDIM * 2;
  float* bsum = (float*)p; p += (size_t)GDIM * 4;
  float* hf   = (float*)p; p += (size_t)BATCH * HDIM * 4;
  float* cf   = (float*)p; p += (size_t)BATCH * HDIM * 4;

  prep_weights<<<4096, 256, 0, stream>>>(w_ih, w_hh, Wc);
  prep_input<<<4096, 256, 0, stream>>>(input, xb);
  prep_state<<<(BATCH * HDIM + 255) / 256, 256, 0, stream>>>(h0, c0, b_ih, b_hh, hb0, cf, bsum);

  for (int t = 0; t < T_STEPS; ++t) {
    const unsigned short* hin = (t & 1) ? hb1 : hb0;  // ping-pong: no cross-block RAW on h
    unsigned short* hout      = (t & 1) ? hb0 : hb1;
    lstm_step<<<HDIM / 32, 256, 0, stream>>>(xb, Wc, bsum, hin, hout, hf, cf, out, t);
  }

  softmax_rows<<<T_STEPS * BATCH, 256, 0, stream>>>(out);
  write_final<<<(BATCH * HDIM + 255) / 256, 256, 0, stream>>>(
      hf, cf, out + (size_t)T_STEPS * BATCH * HDIM);
}
